// S4DBlock_64175401337550
// MI455X (gfx1250) — matrix operand order repaired and checked by reading, not compiled, not run on hardware
//
#include <hip/hip_runtime.h>
#include <math.h>

// ---------------- CDNA5 WMMA types ----------------
typedef __bf16 bf16_t;
typedef bf16_t v16bf __attribute__((ext_vector_type(16)));
typedef float  v8f   __attribute__((ext_vector_type(8)));

#define D_MODEL 512
#define L_SEQ   2048
#define B_SZ    8
#define N_ST    64
#define E_DIM   512
#define LN_EPSF 1e-5f

__device__ __forceinline__ unsigned short f2bf(float f) {
    unsigned int u = __float_as_uint(f);
    u += 0x7FFFu + ((u >> 16) & 1u);          // round-to-nearest-even
    return (unsigned short)(u >> 16);
}
__device__ __forceinline__ float bf2f(unsigned short h) {
    return __uint_as_float(((unsigned int)h) << 16);
}

union FragBF {                                 // 16 bf16 = 8 VGPRs
    v16bf v;
    unsigned short u[16];
    uint4 q[2];
};

// ---------------- prep: W_out f32 -> bf16 ----------------
__global__ void wcvt_kernel(const float* __restrict__ W,
                            unsigned short* __restrict__ Wb, int n) {
    int i = blockIdx.x * blockDim.x + threadIdx.x;
    if (i < n) Wb[i] = f2bf(W[i]);
}

// ---------------- FiLM: silu(emb) @ W_film^T + b_film ----------------
__global__ void film_kernel(const float* __restrict__ emb,
                            const float* __restrict__ Wf,
                            const float* __restrict__ bf,
                            float* __restrict__ film) {
    int b = blockIdx.x;                                    // 0..7
    int j = blockIdx.y * blockDim.x + threadIdx.x;         // 0..1023
    float acc = bf[j];
    const float* e = emb + b * E_DIM;
    const float* w = Wf  + (size_t)j * E_DIM;
    for (int k = 0; k < E_DIM; ++k) {
        float s = e[k];
        acc += (s / (1.f + __expf(-s))) * w[k];
    }
    film[b * 2 * D_MODEL + j] = acc;
}

// ---------------- LayerNorm, write xn^T as bf16 [b][d][l] ----------------
__global__ void ln_kernel(const float* __restrict__ x,
                          const float* __restrict__ gamma,
                          const float* __restrict__ beta,
                          unsigned short* __restrict__ Xt) {
    int row  = blockIdx.x;                  // b*L + l
    int b    = row >> 11;
    int l    = row & (L_SEQ - 1);
    int lane = threadIdx.x;                 // wave32, 16 channels per lane
    const float* xr = x + (size_t)row * D_MODEL;
    float v[16], s = 0.f, s2 = 0.f;
#pragma unroll
    for (int i = 0; i < 16; ++i) {
        float t = xr[lane * 16 + i];
        v[i] = t; s += t; s2 += t * t;
    }
#pragma unroll
    for (int o = 16; o >= 1; o >>= 1) {
        s  += __shfl_xor(s,  o, 32);
        s2 += __shfl_xor(s2, o, 32);
    }
    float mu  = s * (1.f / (float)D_MODEL);
    float var = s2 * (1.f / (float)D_MODEL) - mu * mu;
    float inv = rsqrtf(var + LN_EPSF);
#pragma unroll
    for (int i = 0; i < 16; ++i) {
        int c = lane * 16 + i;
        float xn = (v[i] - mu) * inv * gamma[c] + beta[c];
        Xt[((size_t)(b * D_MODEL + c)) * L_SEQ + l] = f2bf(xn);
    }
}

// ---------------- S4D kernel K[d][l], bf16 ----------------
__global__ void kgen_kernel(const float* __restrict__ Ar,
                            const float* __restrict__ Ai,
                            const float* __restrict__ C,
                            const float* __restrict__ logdt,
                            const float* __restrict__ Bs,
                            unsigned short* __restrict__ Kb) {
    int id = blockIdx.x * blockDim.x + threadIdx.x;   // d*L ids
    int dd = id >> 11;
    int l  = id & (L_SEQ - 1);
    float dt = __expf(logdt[dd]);
    float fl = (float)l;
    float acc = 0.f;
    const float* ar = Ar + dd * N_ST;
    const float* ai = Ai + dd * N_ST;
    const float* cc = C  + (size_t)dd * N_ST * 2;
    const float* bs = Bs + dd * N_ST;
    for (int n = 0; n < N_ST; ++n) {
        float arl = ar[n] * dt * fl;
        float ail = ai[n] * dt * fl;
        float cr  = cc[2 * n + 0] * bs[n];
        float ci  = cc[2 * n + 1] * bs[n];
        float mag = __expf(arl);
        float sn, cs;
        __sincosf(ail, &sn, &cs);
        acc += mag * (cr * cs - ci * sn);
    }
    Kb[id] = f2bf(acc * dt);
}

// ------ causal block-Toeplitz convolution via WMMA + fused D-skip/FiLM ------
// 8 waves per workgroup share one reversed-K LDS image for channel dd.
// Wave w handles block-row i = blockIdx.x*8 + w: 16 timesteps x 16 cols
// (cols 0..7 = batches; cols 8..15 carry duplicate batches and are discarded).
// Inner loop is software-pipelined: next strip's fragments are issued before
// the current WMMA so the wait only covers already-in-flight loads.
__global__ void conv_kernel(const unsigned short* __restrict__ Kb,
                            const unsigned short* __restrict__ Xt,
                            const float* __restrict__ Dskip,
                            const float* __restrict__ film,
                            unsigned short* __restrict__ Ymid) {
    __shared__ unsigned short sK[2080];       // reversed K, +32 zero pad
    const int dd   = blockIdx.y;              // 0..511 channel
    const int tid  = threadIdx.x;
    const int wid  = tid >> 5;                // wave 0..7
    const int lane = tid & 31;
    const int ln16 = lane & 15;
    const int hi   = lane >> 4;
    const int i    = blockIdx.x * 8 + wid;    // block-row 0..127
    const int JmaxMax = blockIdx.x * 128 + 127;

    // Stage reversed, zero-padded K[d]: sK[p] = K[d][JmaxMax-p] (0 if <0)
    const unsigned short* Kr = Kb + (size_t)dd * L_SEQ;
    for (int p = tid; p < JmaxMax + 33; p += 256) {
        int j = JmaxMax - p;
        sK[p] = (j >= 0) ? Kr[j] : (unsigned short)0;
    }
    __syncthreads();

    v8f c;
#pragma unroll
    for (int r = 0; r < 8; ++r) c[r] = 0.f;

    // lanes 8..15 mirror batches 0..7 (results discarded) -> uniform loads
    const unsigned short* Xrow =
        Xt + ((size_t)((ln16 & 7) * D_MODEL + dd)) * L_SEQ;

    const int pbase = JmaxMax - i * 16 - ln16 + hi * 8;   // reversed-K base
    const int nk    = (i >> 1) + 1;                       // 32-wide K strips

    FragBF a0, b0, a1, b1;
    auto loadA = [&](FragBF& f, int kt) {
        int p0 = pbase + kt * 32;
#pragma unroll
        for (int w = 0; w < 8; ++w) f.u[w]     = sK[p0 + w];
#pragma unroll
        for (int w = 0; w < 8; ++w) f.u[8 + w] = sK[p0 + 16 + w];
    };
    auto loadB = [&](FragBF& f, int kt) {
        const uint4* pb = (const uint4*)(Xrow + kt * 32 + hi * 8);
        f.q[0] = pb[0];
        f.q[1] = pb[2];
    };

    loadA(a0, 0);
    loadB(b0, 0);
    int kt = 0;
    while (true) {
        int ktn = (kt + 1 < nk) ? kt + 1 : kt;   // uniform clamp, no branch
        loadA(a1, ktn);
        loadB(b1, ktn);
        c = __builtin_amdgcn_wmma_f32_16x16x32_bf16(
                false, a0.v, false, b0.v, (short)0, c, false, false);
        if (++kt >= nk) break;
        ktn = (kt + 1 < nk) ? kt + 1 : kt;
        loadA(a0, ktn);
        loadB(b0, ktn);
        c = __builtin_amdgcn_wmma_f32_16x16x32_bf16(
                false, a1.v, false, b1.v, (short)0, c, false, false);
        if (++kt >= nk) break;
    }

    // Fused epilogue: + xn*D_skip, FiLM scale/shift, write y_mid bf16 [b][l][d]
    if (ln16 < 8) {
        float ds = Dskip[dd];
        float sc = film[ln16 * 2 * D_MODEL + dd];
        float sh = film[ln16 * 2 * D_MODEL + D_MODEL + dd];
#pragma unroll
        for (int r = 0; r < 8; ++r) {
            int l = i * 16 + r + hi * 8;       // D layout: M = r + 8*(lane/16)
            float xn = bf2f(Xrow[l]);
            float y  = c[r] + xn * ds;
            y = y * (1.f + sc) + sh;
            Ymid[((size_t)(ln16 * L_SEQ + l)) * D_MODEL + dd] = f2bf(y);
        }
    }
}

// ------ out_proj GEMM (M=16384,K=512,N=1024) with fused GLU + residual ------
// One wave per (16-token tile, 32-col tile): 2 "a" tiles + matching 2 "g"
// tiles (rows +512) -> 4 WMMAs sharing each A fragment. K loop is
// software-pipelined with ping-pong fragment sets.
__global__ void outproj_kernel(const unsigned short* __restrict__ Ymid,
                               const unsigned short* __restrict__ Wb,
                               const float* __restrict__ bout,
                               const float* __restrict__ x,
                               float* __restrict__ out) {
    const int t0   = blockIdx.x * 16;         // token tile
    const int j0   = blockIdx.y * 32;         // output-col tile (0..511, 2x16)
    const int lane = threadIdx.x;
    const int ln16 = lane & 15;
    const int hi   = lane >> 4;

    v8f ca0, ca1, cg0, cg1;
#pragma unroll
    for (int r = 0; r < 8; ++r) { ca0[r] = 0.f; ca1[r] = 0.f;
                                  cg0[r] = 0.f; cg1[r] = 0.f; }

    const unsigned short* Arow = Ymid + (size_t)(t0 + ln16) * D_MODEL;
    const unsigned short* Wa0  = Wb + (size_t)(j0 + ln16) * D_MODEL;
    const unsigned short* Wa1  = Wb + (size_t)(j0 + 16 + ln16) * D_MODEL;
    const unsigned short* Wg0  = Wb + (size_t)(j0 + D_MODEL + ln16) * D_MODEL;
    const unsigned short* Wg1  = Wb + (size_t)(j0 + D_MODEL + 16 + ln16) * D_MODEL;

    struct Set { FragBF a, wa0, wa1, wg0, wg1; };
    Set s0, s1;
    auto loadSet = [&](Set& s, int k0) {
        s.a.q[0]   = *(const uint4*)(Arow + k0 + hi * 16);
        s.a.q[1]   = *(const uint4*)(Arow + k0 + 8 + hi * 16);
        const uint4* p0 = (const uint4*)(Wa0 + k0 + hi * 16);
        s.wa0.q[0] = p0[0]; s.wa0.q[1] = p0[1];
        const uint4* p1 = (const uint4*)(Wa1 + k0 + hi * 16);
        s.wa1.q[0] = p1[0]; s.wa1.q[1] = p1[1];
        const uint4* p2 = (const uint4*)(Wg0 + k0 + hi * 16);
        s.wg0.q[0] = p2[0]; s.wg0.q[1] = p2[1];
        const uint4* p3 = (const uint4*)(Wg1 + k0 + hi * 16);
        s.wg1.q[0] = p3[0]; s.wg1.q[1] = p3[1];
    };
    auto mmaSet = [&](Set& s) {
        ca0 = __builtin_amdgcn_wmma_f32_16x16x32_bf16(
                  false, s.a.v, false, s.wa0.v, (short)0, ca0, false, false);
        ca1 = __builtin_amdgcn_wmma_f32_16x16x32_bf16(
                  false, s.a.v, false, s.wa1.v, (short)0, ca1, false, false);
        cg0 = __builtin_amdgcn_wmma_f32_16x16x32_bf16(
                  false, s.a.v, false, s.wg0.v, (short)0, cg0, false, false);
        cg1 = __builtin_amdgcn_wmma_f32_16x16x32_bf16(
                  false, s.a.v, false, s.wg1.v, (short)0, cg1, false, false);
    };

    loadSet(s0, 0);
#pragma unroll
    for (int k0 = 0; k0 < D_MODEL; k0 += 64) {
        loadSet(s1, k0 + 32);                  // prefetch odd strip
        mmaSet(s0);
        int kn = (k0 + 64 < D_MODEL) ? k0 + 64 : k0;  // clamp, stays in bounds
        loadSet(s0, kn);                       // prefetch next even strip
        mmaSet(s1);
    }

    float ba0 = bout[j0 + ln16];
    float ba1 = bout[j0 + 16 + ln16];
    float bg0 = bout[j0 + D_MODEL + ln16];
    float bg1 = bout[j0 + D_MODEL + 16 + ln16];
#pragma unroll
    for (int r = 0; r < 8; ++r) {
        int t = t0 + r + hi * 8;
        size_t row = (size_t)t * D_MODEL;
        {
            size_t idx = row + (j0 + ln16);
            float aa  = ca0[r] + ba0;
            float gg  = cg0[r] + bg0;
            out[idx]  = x[idx] + aa * (1.f / (1.f + __expf(-gg)));
        }
        {
            size_t idx = row + (j0 + 16 + ln16);
            float aa  = ca1[r] + ba1;
            float gg  = cg1[r] + bg1;
            out[idx]  = x[idx] + aa * (1.f / (1.f + __expf(-gg)));
        }
    }
}

// ---------------- launch ----------------
extern "C" void kernel_launch(void* const* d_in, const int* in_sizes, int n_in,
                              void* d_out, int out_size, void* d_ws, size_t ws_size,
                              hipStream_t stream) {
    const float* x      = (const float*)d_in[0];
    const float* emb    = (const float*)d_in[1];
    const float* A_real = (const float*)d_in[2];
    const float* A_imag = (const float*)d_in[3];
    const float* C      = (const float*)d_in[4];
    const float* log_dt = (const float*)d_in[5];
    const float* B_ssm  = (const float*)d_in[6];
    const float* D_skip = (const float*)d_in[7];
    const float* ln_g   = (const float*)d_in[8];
    const float* ln_b   = (const float*)d_in[9];
    const float* W_out  = (const float*)d_in[10];
    const float* b_out  = (const float*)d_in[11];
    const float* W_film = (const float*)d_in[12];
    const float* b_film = (const float*)d_in[13];
    float* out = (float*)d_out;

    // workspace layout (bytes), ~36.7 MB total
    char* ws = (char*)d_ws;
    unsigned short* Xt   = (unsigned short*)(ws + 0);          // 16,777,216  xn^T bf16 [b][d][l]
    unsigned short* Kb   = (unsigned short*)(ws + 16777216);   //  2,097,152  K bf16 [d][l]
    unsigned short* Wb   = (unsigned short*)(ws + 18874368);   //  1,048,576  W_out bf16
    float*          film = (float*)        (ws + 19922944);    //     32,768  film f32 [b][2d]
    unsigned short* Ymid = (unsigned short*)(ws + 19955712);   // 16,777,216  y bf16 [b][l][d]

    wcvt_kernel<<<(2 * D_MODEL * D_MODEL + 255) / 256, 256, 0, stream>>>(
        W_out, Wb, 2 * D_MODEL * D_MODEL);
    film_kernel<<<dim3(B_SZ, (2 * D_MODEL) / 256), 256, 0, stream>>>(
        emb, W_film, b_film, film);
    ln_kernel<<<B_SZ * L_SEQ, 32, 0, stream>>>(x, ln_g, ln_b, Xt);
    kgen_kernel<<<(D_MODEL * L_SEQ) / 256, 256, 0, stream>>>(
        A_real, A_imag, C, log_dt, B_ssm, Kb);
    conv_kernel<<<dim3(L_SEQ / 16 / 8, D_MODEL), 256, 0, stream>>>(
        Kb, Xt, D_skip, film, Ymid);
    outproj_kernel<<<dim3((B_SZ * L_SEQ) / 16, D_MODEL / 32), 32, 0, stream>>>(
        Ymid, Wb, b_out, x, out);
}
